// MoDBlock_47571057770501
// MI455X (gfx1250) — matrix operand order repaired and checked by reading, not compiled, not run on hardware
//
#include <hip/hip_runtime.h>
#include <math.h>

#define B_    2
#define S_    2048
#define D_    2048
#define H_    16
#define HD_   128
#define TOPK_ 256
#define HID_  8192
#define NT_   (B_*S_)
#define EPS_  1e-6f

typedef float  v8f   __attribute__((ext_vector_type(8)));
typedef __bf16 v16bf __attribute__((ext_vector_type(16)));
typedef __bf16 v8bf  __attribute__((ext_vector_type(8)));
typedef int    v4i_  __attribute__((vector_size(16)));

// ---- CDNA5 async global->LDS staging (ASYNCcnt) with safe fallback ----
#if defined(__has_builtin)
#if __has_builtin(__builtin_amdgcn_global_load_async_to_lds_b128)
#define HAS_ASYNC_LDS 1
#endif
#endif
#ifndef HAS_ASYNC_LDS
#define HAS_ASYNC_LDS 0
#endif

static __device__ __forceinline__ void stage16(const __bf16* g, __bf16* l) {
#if HAS_ASYNC_LDS
  __builtin_amdgcn_global_load_async_to_lds_b128(
      (__attribute__((address_space(1))) v4i_*)g,
      (__attribute__((address_space(3))) v4i_*)l, 0, 0);
#else
  *reinterpret_cast<v8bf*>(l) = *reinterpret_cast<const v8bf*>(g);
#endif
}
static __device__ __forceinline__ void stage_wait() {
#if HAS_ASYNC_LDS
#if __has_builtin(__builtin_amdgcn_s_wait_asynccnt)
  __builtin_amdgcn_s_wait_asynccnt(0);
#else
  asm volatile("s_wait_asynccnt 0x0" ::: "memory");
#endif
#endif
}

static __device__ __forceinline__ float warp_sum(float v) {
  v += __shfl_xor(v, 1, 32);  v += __shfl_xor(v, 2, 32);
  v += __shfl_xor(v, 4, 32);  v += __shfl_xor(v, 8, 32);
  v += __shfl_xor(v, 16, 32);
  return v;
}

// ---------------- router: token_weights = x @ router_w ----------------
__global__ __launch_bounds__(256) void k_router(const float* __restrict__ x,
                                                const float* __restrict__ rw,
                                                float* __restrict__ tw) {
  int t = blockIdx.x;
  const float4* xr = reinterpret_cast<const float4*>(x + (size_t)t * D_);
  const float4* wr = reinterpret_cast<const float4*>(rw);
  float acc = 0.f;
  for (int i = threadIdx.x; i < D_ / 4; i += 256) {
    float4 a = xr[i], b = wr[i];
    acc += a.x * b.x + a.y * b.y + a.z * b.z + a.w * b.w;
  }
  acc = warp_sum(acc);
  __shared__ float sred[8];
  if ((threadIdx.x & 31) == 0) sred[threadIdx.x >> 5] = acc;
  __syncthreads();
  if (threadIdx.x == 0) {
    float s = 0.f;
    for (int i = 0; i < 8; i++) s += sred[i];
    tw[t] = s;
  }
}

// ---------------- top-k via LDS bitonic sort (desc value, asc idx) ----
__global__ __launch_bounds__(1024) void k_topk(const float* __restrict__ tw,
                                               int* __restrict__ idx_out,
                                               float* __restrict__ mask) {
  int b = blockIdx.x, tid = threadIdx.x;
  __shared__ float sv[S_];
  __shared__ int   si[S_];
  for (int i = tid; i < S_; i += 1024) {
    sv[i] = tw[(size_t)b * S_ + i];
    si[i] = i;
    mask[(size_t)b * S_ + i] = 0.f;
  }
  __syncthreads();
  for (int k = 2; k <= S_; k <<= 1) {
    for (int j = k >> 1; j > 0; j >>= 1) {
      for (int i = tid; i < S_; i += 1024) {
        int ixj = i ^ j;
        if (ixj > i) {
          float v0 = sv[i], v1 = sv[ixj];
          int   i0 = si[i], i1 = si[ixj];
          bool less01 = (v0 > v1) || (v0 == v1 && i0 < i1);
          bool asc = ((i & k) == 0);
          bool sw = asc ? !less01 : less01;
          if (sw) { sv[i] = v1; sv[ixj] = v0; si[i] = i1; si[ixj] = i0; }
        }
      }
      __syncthreads();
    }
  }
  if (tid < TOPK_) {
    int id = si[tid];
    idx_out[b * TOPK_ + tid] = id;
    mask[(size_t)b * S_ + id] = 1.f;
  }
}

// ---------------- fp32 RxC -> bf16 CxR (tiled transpose-convert) ------
__global__ __launch_bounds__(256) void k_cvt_t(const float* __restrict__ src,
                                               __bf16* __restrict__ dst,
                                               int R, int C) {
  __shared__ __bf16 tile[32][33];
  int bx = blockIdx.x * 32;            // source column base
  int by = blockIdx.y * 32;            // source row base
  int tx = threadIdx.x & 31, ty = threadIdx.x >> 5;
#pragma unroll
  for (int i = 0; i < 32; i += 8)
    tile[ty + i][tx] = (__bf16)src[(size_t)(by + ty + i) * C + bx + tx];
  __syncthreads();
#pragma unroll
  for (int i = 0; i < 32; i += 8)
    dst[(size_t)(bx + ty + i) * R + by + tx] = tile[tx][ty + i];
}

// -------- V (B,S,H,HD) fp32 -> Vt (B,H,HD,S) bf16 (tiled) -------------
__global__ __launch_bounds__(256) void k_cvt_vt(const float* __restrict__ src,
                                                __bf16* __restrict__ dst) {
  __shared__ __bf16 tile[32][33];
  int bh = blockIdx.z;
  int b = bh >> 4, h = bh & 15;
  int s0 = blockIdx.y * 32, d0 = blockIdx.x * 32;
  int tx = threadIdx.x & 31, ty = threadIdx.x >> 5;
#pragma unroll
  for (int i = 0; i < 32; i += 8)
    tile[ty + i][tx] =
        (__bf16)src[((size_t)b * S_ + s0 + ty + i) * (H_ * HD_) + (size_t)h * HD_ + d0 + tx];
  __syncthreads();
#pragma unroll
  for (int i = 0; i < 32; i += 8)
    dst[((size_t)bh * HD_ + d0 + ty + i) * S_ + s0 + tx] = tile[tx][ty + i];
}

// ---------------- masked input + RMSNorm -> bf16 ----------------------
__global__ __launch_bounds__(256) void k_norm_in(const float* __restrict__ x,
                                                 const float* __restrict__ mask,
                                                 const float* __restrict__ nw,
                                                 float* __restrict__ masked,
                                                 __bf16* __restrict__ xn) {
  int t = blockIdx.x;
  float m = mask[t];
  const float* xr = x + (size_t)t * D_;
  float* mr = masked + (size_t)t * D_;
  __bf16* nr = xn + (size_t)t * D_;
  float vals[8], ss = 0.f;
#pragma unroll
  for (int i = 0; i < 8; i++) {
    int d = threadIdx.x + i * 256;
    float v = xr[d] * m;
    vals[i] = v; mr[d] = v; ss += v * v;
  }
  ss = warp_sum(ss);
  __shared__ float sred[8];
  if ((threadIdx.x & 31) == 0) sred[threadIdx.x >> 5] = ss;
  __syncthreads();
  float tot = 0.f;
  for (int i = 0; i < 8; i++) tot += sred[i];
  float rinv = rsqrtf(tot / D_ + EPS_);
#pragma unroll
  for (int i = 0; i < 8; i++) {
    int d = threadIdx.x + i * 256;
    nr[d] = (__bf16)(vals[i] * rinv * nw[d]);
  }
}

// ---------------- h = a + b; RMSNorm -> bf16 --------------------------
__global__ __launch_bounds__(256) void k_resid_norm(const float* __restrict__ a,
                                                    const float* __restrict__ bv,
                                                    const float* __restrict__ nw,
                                                    __bf16* __restrict__ hn) {
  int t = blockIdx.x;
  const float* ar = a + (size_t)t * D_;
  const float* br = bv + (size_t)t * D_;
  __bf16* nr = hn + (size_t)t * D_;
  float vals[8], ss = 0.f;
#pragma unroll
  for (int i = 0; i < 8; i++) {
    int d = threadIdx.x + i * 256;
    float v = ar[d] + br[d];
    vals[i] = v; ss += v * v;
  }
  ss = warp_sum(ss);
  __shared__ float sred[8];
  if ((threadIdx.x & 31) == 0) sred[threadIdx.x >> 5] = ss;
  __syncthreads();
  float tot = 0.f;
  for (int i = 0; i < 8; i++) tot += sred[i];
  float rinv = rsqrtf(tot / D_ + EPS_);
#pragma unroll
  for (int i = 0; i < 8; i++) {
    int d = threadIdx.x + i * 256;
    nr[d] = (__bf16)(vals[i] * rinv * nw[d]);
  }
}

// --- bf16 WMMA GEMM: C[M,N] = A[M,K] * W[K,N], Bt = W^T stored [N,K] --
// 128x128x32 tile, 8 waves (2x4), 64x32 per wave, double-buffered LDS.
#define ASTR 40
__global__ __launch_bounds__(256) void k_gemm(const __bf16* __restrict__ A,
                                              const __bf16* __restrict__ Bt,
                                              float* __restrict__ C,
                                              int M, int N, int K) {
  __shared__ __bf16 sA[2][128 * ASTR];
  __shared__ __bf16 sB[2][128 * ASTR];
  const int tid = threadIdx.x, lane = tid & 31, wid = tid >> 5;
  const int lr = lane & 15, hi = lane >> 4;
  const int wm = wid & 1, wn = wid >> 1;
  const int m0 = blockIdx.y * 128, n0 = blockIdx.x * 128;
  const int sr = tid >> 1, scol = (tid & 1) * 16;   // staging: row 0..127, col 0/16
  v8f acc[4][2] = {};

  const __bf16* gA = A + (size_t)(m0 + sr) * K + scol;
  const __bf16* gB = Bt + (size_t)(n0 + sr) * K + scol;
  // prologue: stage tile 0
  stage16(gA,     &sA[0][sr * ASTR + scol]);
  stage16(gA + 8, &sA[0][sr * ASTR + scol + 8]);
  stage16(gB,     &sB[0][sr * ASTR + scol]);
  stage16(gB + 8, &sB[0][sr * ASTR + scol + 8]);

  int cur = 0;
  for (int k0 = 0; k0 < K; k0 += 32) {
    stage_wait();
    __syncthreads();
    if (k0 + 32 < K) {   // stage next tile into the other buffer
      const __bf16* nA = gA + k0 + 32;
      const __bf16* nB = gB + k0 + 32;
      int nb = cur ^ 1;
      stage16(nA,     &sA[nb][sr * ASTR + scol]);
      stage16(nA + 8, &sA[nb][sr * ASTR + scol + 8]);
      stage16(nB,     &sB[nb][sr * ASTR + scol]);
      stage16(nB + 8, &sB[nb][sr * ASTR + scol + 8]);
    }
    v16bf af[4];
#pragma unroll
    for (int i = 0; i < 4; i++) {
      int row = wm * 64 + i * 16 + lr;
      v8bf* ap = reinterpret_cast<v8bf*>(&af[i]);
      ap[0] = *reinterpret_cast<v8bf*>(&sA[cur][row * ASTR + hi * 8]);
      ap[1] = *reinterpret_cast<v8bf*>(&sA[cur][row * ASTR + 16 + hi * 8]);
    }
    v16bf bfr[2];
#pragma unroll
    for (int j = 0; j < 2; j++) {
      int col = wn * 32 + j * 16 + lr;
      v8bf* bp = reinterpret_cast<v8bf*>(&bfr[j]);
      bp[0] = *reinterpret_cast<v8bf*>(&sB[cur][col * ASTR + hi * 8]);
      bp[1] = *reinterpret_cast<v8bf*>(&sB[cur][col * ASTR + 16 + hi * 8]);
    }
#pragma unroll
    for (int i = 0; i < 4; i++)
#pragma unroll
      for (int j = 0; j < 2; j++)
        acc[i][j] = __builtin_amdgcn_wmma_f32_16x16x32_bf16(
            false, af[i], false, bfr[j], (short)0, acc[i][j], false, false);
    cur ^= 1;
  }
#pragma unroll
  for (int i = 0; i < 4; i++) {
    int rbase = m0 + wm * 64 + i * 16 + hi * 8;
#pragma unroll
    for (int j = 0; j < 2; j++) {
      int col = n0 + wn * 32 + j * 16 + lr;
#pragma unroll
      for (int g = 0; g < 8; g++)
        C[(size_t)(rbase + g) * N + col] = acc[i][j][g];
    }
  }
}

// ---------------- RoPE (fp32 in -> bf16 out) --------------------------
__global__ void k_rotary(const float* __restrict__ in, const float* __restrict__ fr,
                         __bf16* __restrict__ out) {
  int t = blockIdx.x, h = blockIdx.y, p = threadIdx.x;
  int s = t & (S_ - 1);
  float c  = fr[(s * 64 + p) * 2];
  float sn = fr[(s * 64 + p) * 2 + 1];
  size_t base = (size_t)t * (H_ * HD_) + (size_t)h * HD_ + 2 * p;
  float a = in[base], b = in[base + 1];
  out[base]     = (__bf16)(a * c - b * sn);
  out[base + 1] = (__bf16)(a * sn + b * c);
}

// ---------------- flash attention with WMMA ---------------------------
// Q,K: (B,S,H,HD) bf16 row-major; Vt: (B,H,HD,S) bf16 (pre-transposed)
#define KSTR 136
#define VSTR 40
#define PSTR 40
__global__ __launch_bounds__(256) void k_attn(const __bf16* __restrict__ Q,
                                              const __bf16* __restrict__ Kb,
                                              const __bf16* __restrict__ Vt,
                                              const float* __restrict__ mask,
                                              __bf16* __restrict__ O) {
  __shared__ __bf16 sK[32 * KSTR];
  __shared__ __bf16 sVt[128 * VSTR];
  __shared__ __bf16 sP[8 * 16 * PSTR];
  __shared__ float smask[32];
  const int tid = threadIdx.x, lane = tid & 31, wid = tid >> 5;
  const int lr = lane & 15, hi = lane >> 4;
  const int q0 = blockIdx.x * 128, h = blockIdx.y, b = blockIdx.z;
  const size_t rs = (size_t)(H_ * HD_);
  const size_t base = ((size_t)b * S_) * rs + (size_t)h * HD_;
  const size_t vtb = ((size_t)(b * H_ + h)) * HD_ * S_;

  v16bf qf[4];
  {
    size_t qrow = base + (size_t)(q0 + wid * 16 + lr) * rs;
#pragma unroll
    for (int kf = 0; kf < 4; kf++) {
      v8bf* qp = reinterpret_cast<v8bf*>(&qf[kf]);
      qp[0] = *reinterpret_cast<const v8bf*>(&Q[qrow + kf * 32 + hi * 8]);
      qp[1] = *reinterpret_cast<const v8bf*>(&Q[qrow + kf * 32 + 16 + hi * 8]);
    }
  }
  float mrow[8], lrow[8];
#pragma unroll
  for (int g = 0; g < 8; g++) { mrow[g] = -3.0e38f; lrow[g] = 0.f; }
  v8f o[8] = {};
  const float scale = 0.08838834764831845f;  // 1/sqrt(128)
  __bf16* sPw = &sP[wid * 16 * PSTR];
  const int kk = tid >> 3, kc = (tid & 7) * 16;   // K staging: 32 rows x 128
  const int vr2 = tid >> 1, vc = (tid & 1) * 16;  // Vt staging: 128 rows x 32

  for (int kt = 0; kt < S_ / 32; kt++) {
    int kbase = kt * 32;
    {   // stage K tile (row-major) and Vt tile (row-major, pre-transposed)
      const __bf16* gk = &Kb[base + (size_t)(kbase + kk) * rs + kc];
      stage16(gk,     &sK[kk * KSTR + kc]);
      stage16(gk + 8, &sK[kk * KSTR + kc + 8]);
      const __bf16* gv = &Vt[vtb + (size_t)vr2 * S_ + kbase + vc];
      stage16(gv,     &sVt[vr2 * VSTR + vc]);
      stage16(gv + 8, &sVt[vr2 * VSTR + vc + 8]);
      if (tid < 32) smask[tid] = mask[(size_t)b * S_ + kbase + tid];
    }
    stage_wait();
    __syncthreads();
    v8f sc[2] = {};
#pragma unroll
    for (int j = 0; j < 2; j++)
#pragma unroll
      for (int kf = 0; kf < 4; kf++) {
        v16bf kfr;
        v8bf* kp = reinterpret_cast<v8bf*>(&kfr);
        int krow = j * 16 + lr;
        kp[0] = *reinterpret_cast<v8bf*>(&sK[krow * KSTR + kf * 32 + hi * 8]);
        kp[1] = *reinterpret_cast<v8bf*>(&sK[krow * KSTR + kf * 32 + 16 + hi * 8]);
        sc[j] = __builtin_amdgcn_wmma_f32_16x16x32_bf16(
            false, qf[kf], false, kfr, (short)0, sc[j], false, false);
      }
    float bias0 = smask[lr] > 0.f ? 0.f : -1.0e9f;
    float bias1 = smask[16 + lr] > 0.f ? 0.f : -1.0e9f;
    float co[8];
#pragma unroll
    for (int g = 0; g < 8; g++) {
      float a0 = sc[0][g] * scale + bias0;
      float a1 = sc[1][g] * scale + bias1;
      float t = fmaxf(a0, a1);
      t = fmaxf(t, __shfl_xor(t, 1, 32));
      t = fmaxf(t, __shfl_xor(t, 2, 32));
      t = fmaxf(t, __shfl_xor(t, 4, 32));
      t = fmaxf(t, __shfl_xor(t, 8, 32));
      float newm = fmaxf(mrow[g], t);
      co[g] = __expf(mrow[g] - newm);
      float p0 = __expf(a0 - newm), p1 = __expf(a1 - newm);
      sc[0][g] = p0; sc[1][g] = p1;
      float rsum = p0 + p1;
      rsum += __shfl_xor(rsum, 1, 32);
      rsum += __shfl_xor(rsum, 2, 32);
      rsum += __shfl_xor(rsum, 4, 32);
      rsum += __shfl_xor(rsum, 8, 32);
      lrow[g] = lrow[g] * co[g] + rsum;
      mrow[g] = newm;
    }
#pragma unroll
    for (int n = 0; n < 8; n++)
#pragma unroll
      for (int g = 0; g < 8; g++) o[n][g] *= co[g];
    // transpose P (C-layout) -> A-fragment layout via per-wave LDS
#pragma unroll
    for (int j = 0; j < 2; j++)
#pragma unroll
      for (int g = 0; g < 8; g++)
        sPw[(g + hi * 8) * PSTR + j * 16 + lr] = (__bf16)sc[j][g];
    __syncthreads();
    v16bf pf;
    {
      v8bf* pp = reinterpret_cast<v8bf*>(&pf);
      pp[0] = *reinterpret_cast<v8bf*>(&sPw[lr * PSTR + hi * 8]);
      pp[1] = *reinterpret_cast<v8bf*>(&sPw[lr * PSTR + 16 + hi * 8]);
    }
#pragma unroll
    for (int n = 0; n < 8; n++) {
      v16bf vfr;
      v8bf* vp = reinterpret_cast<v8bf*>(&vfr);
      int vr = n * 16 + lr;
      vp[0] = *reinterpret_cast<v8bf*>(&sVt[vr * VSTR + hi * 8]);
      vp[1] = *reinterpret_cast<v8bf*>(&sVt[vr * VSTR + 16 + hi * 8]);
      o[n] = __builtin_amdgcn_wmma_f32_16x16x32_bf16(
          false, pf, false, vfr, (short)0, o[n], false, false);
    }
    __syncthreads();
  }
  {
    size_t orow = base + (size_t)(q0 + wid * 16 + hi * 8) * rs;
#pragma unroll
    for (int g = 0; g < 8; g++) {
      float rl = 1.f / lrow[g];
#pragma unroll
      for (int n = 0; n < 8; n++)
        O[orow + (size_t)g * rs + n * 16 + lr] = (__bf16)(o[n][g] * rl);
    }
  }
}

// ---- fused SwiGLU gate: silu(A*W1) * (A*W3) -> bf16 ------------------
// W1t/W3t stored transposed [HID, D]; 128x64x32 tile, double-buffered.
__global__ __launch_bounds__(256) void k_ffn_gate(const __bf16* __restrict__ A,
                                                  const __bf16* __restrict__ W1t,
                                                  const __bf16* __restrict__ W3t,
                                                  __bf16* __restrict__ A2) {
  __shared__ __bf16 sA[2][128 * ASTR];
  __shared__ __bf16 sB1[2][64 * ASTR];
  __shared__ __bf16 sB3[2][64 * ASTR];
  const int tid = threadIdx.x, lane = tid & 31, wid = tid >> 5;
  const int lr = lane & 15, hi = lane >> 4;
  const int wm = wid & 1, wn = wid >> 1;
  const int m0 = blockIdx.y * 128, n0 = blockIdx.x * 64;
  const int sr = tid >> 1, scol = (tid & 1) * 16;  // A staging
  const int br = tid >> 2, bc = (tid & 3) * 8;     // B staging: 64 rows x 32
  v8f a1[4] = {}, a3[4] = {};

  const __bf16* gA = A + (size_t)(m0 + sr) * D_ + scol;
  const __bf16* g1 = W1t + (size_t)(n0 + br) * D_ + bc;
  const __bf16* g3 = W3t + (size_t)(n0 + br) * D_ + bc;
  stage16(gA,     &sA[0][sr * ASTR + scol]);
  stage16(gA + 8, &sA[0][sr * ASTR + scol + 8]);
  stage16(g1, &sB1[0][br * ASTR + bc]);
  stage16(g3, &sB3[0][br * ASTR + bc]);

  int cur = 0;
  for (int k0 = 0; k0 < D_; k0 += 32) {
    stage_wait();
    __syncthreads();
    if (k0 + 32 < D_) {
      int nb = cur ^ 1;
      stage16(gA + k0 + 32,     &sA[nb][sr * ASTR + scol]);
      stage16(gA + k0 + 40,     &sA[nb][sr * ASTR + scol + 8]);
      stage16(g1 + k0 + 32, &sB1[nb][br * ASTR + bc]);
      stage16(g3 + k0 + 32, &sB3[nb][br * ASTR + bc]);
    }
    v16bf af[4];
#pragma unroll
    for (int i = 0; i < 4; i++) {
      int row = wm * 64 + i * 16 + lr;
      v8bf* ap = reinterpret_cast<v8bf*>(&af[i]);
      ap[0] = *reinterpret_cast<v8bf*>(&sA[cur][row * ASTR + hi * 8]);
      ap[1] = *reinterpret_cast<v8bf*>(&sA[cur][row * ASTR + 16 + hi * 8]);
    }
    v16bf b1f, b3f;
    {
      int col = wn * 16 + lr;
      v8bf* p1 = reinterpret_cast<v8bf*>(&b1f);
      p1[0] = *reinterpret_cast<v8bf*>(&sB1[cur][col * ASTR + hi * 8]);
      p1[1] = *reinterpret_cast<v8bf*>(&sB1[cur][col * ASTR + 16 + hi * 8]);
      v8bf* p3 = reinterpret_cast<v8bf*>(&b3f);
      p3[0] = *reinterpret_cast<v8bf*>(&sB3[cur][col * ASTR + hi * 8]);
      p3[1] = *reinterpret_cast<v8bf*>(&sB3[cur][col * ASTR + 16 + hi * 8]);
    }
#pragma unroll
    for (int i = 0; i < 4; i++) {
      a1[i] = __builtin_amdgcn_wmma_f32_16x16x32_bf16(false, af[i], false, b1f,
                                                      (short)0, a1[i], false, false);
      a3[i] = __builtin_amdgcn_wmma_f32_16x16x32_bf16(false, af[i], false, b3f,
                                                      (short)0, a3[i], false, false);
    }
    cur ^= 1;
  }
#pragma unroll
  for (int i = 0; i < 4; i++) {
    int rbase = m0 + wm * 64 + i * 16 + hi * 8;
    int col = n0 + wn * 16 + lr;
#pragma unroll
    for (int g = 0; g < 8; g++) {
      float c1 = a1[i][g], c3 = a3[i][g];
      float sig = 1.f / (1.f + __expf(-c1));
      A2[(size_t)(rbase + g) * HID_ + col] = (__bf16)(c1 * sig * c3);
    }
  }
}

// ======================= host side ====================================
extern "C" void kernel_launch(void* const* d_in, const int* in_sizes, int n_in,
                              void* d_out, int out_size, void* d_ws, size_t ws_size,
                              hipStream_t stream) {
  (void)in_sizes; (void)n_in; (void)out_size; (void)ws_size;
  const float* x   = (const float*)d_in[0];
  const float* fr  = (const float*)d_in[2];
  const float* rw  = (const float*)d_in[3];
  const float* wq  = (const float*)d_in[4];
  const float* wk  = (const float*)d_in[5];
  const float* wv  = (const float*)d_in[6];
  const float* wo  = (const float*)d_in[7];
  const float* w1  = (const float*)d_in[8];
  const float* w2  = (const float*)d_in[9];
  const float* w3  = (const float*)d_in[10];
  const float* anw = (const float*)d_in[11];
  const float* fnw = (const float*)d_in[12];

  float* out_p = (float*)d_out;
  float* tw_p  = out_p + (size_t)NT_ * D_;
  int*   idx_p = (int*)(tw_p + NT_);

  char* wsb = (char*)d_ws;
  size_t off = 0;
  auto alloc = [&](size_t bytes) -> void* {
    void* p = wsb + off;
    off = (off + bytes + 255) & ~(size_t)255;
    return p;
  };
  const size_t NQK = (size_t)D_ * H_ * HD_;       // 4M
  const size_t NW  = (size_t)D_ * HID_;           // 16M
  const size_t NTD = (size_t)NT_ * D_;            // 8M
  __bf16* wqt = (__bf16*)alloc(NQK * 2);          // all weights transposed [N,K]
  __bf16* wkt = (__bf16*)alloc(NQK * 2);
  __bf16* wvt = (__bf16*)alloc(NQK * 2);
  __bf16* wot = (__bf16*)alloc(NQK * 2);
  __bf16* w1t = (__bf16*)alloc(NW * 2);
  __bf16* w2t = (__bf16*)alloc(NW * 2);
  __bf16* w3t = (__bf16*)alloc(NW * 2);
  float*  maskp   = (float*)alloc((size_t)NT_ * 4);
  float*  maskedp = (float*)alloc(NTD * 4);
  __bf16* xn   = (__bf16*)alloc(NTD * 2);
  float*  qf   = (float*)alloc(NTD * 4);
  float*  kf_  = (float*)alloc(NTD * 4);
  float*  vf   = (float*)alloc(NTD * 4);
  __bf16* qb   = (__bf16*)alloc(NTD * 2);
  __bf16* kb   = (__bf16*)alloc(NTD * 2);
  __bf16* vt   = (__bf16*)alloc(NTD * 2);         // (B,H,HD,S)
  __bf16* ob   = (__bf16*)alloc(NTD * 2);
  float*  aproj = (float*)alloc(NTD * 4);
  __bf16* hn   = (__bf16*)alloc(NTD * 2);
  __bf16* a2   = (__bf16*)alloc((size_t)NT_ * HID_ * 2);

  // 1) router + top-k + mask
  k_router<<<NT_, 256, 0, stream>>>(x, rw, tw_p);
  k_topk<<<B_, 1024, 0, stream>>>(tw_p, idx_p, maskp);

  // 2) weights fp32 -> bf16, transposed (coalesced tiled transpose)
  k_cvt_t<<<dim3(D_ / 32, D_ / 32), 256, 0, stream>>>(wq, wqt, D_, D_);
  k_cvt_t<<<dim3(D_ / 32, D_ / 32), 256, 0, stream>>>(wk, wkt, D_, D_);
  k_cvt_t<<<dim3(D_ / 32, D_ / 32), 256, 0, stream>>>(wv, wvt, D_, D_);
  k_cvt_t<<<dim3(D_ / 32, D_ / 32), 256, 0, stream>>>(wo, wot, D_, D_);
  k_cvt_t<<<dim3(HID_ / 32, D_ / 32), 256, 0, stream>>>(w1, w1t, D_, HID_);
  k_cvt_t<<<dim3(D_ / 32, HID_ / 32), 256, 0, stream>>>(w2, w2t, HID_, D_);
  k_cvt_t<<<dim3(HID_ / 32, D_ / 32), 256, 0, stream>>>(w3, w3t, D_, HID_);

  // 3) masked input + attn RMSNorm
  k_norm_in<<<NT_, 256, 0, stream>>>(x, maskp, anw, maskedp, xn);

  // 4) QKV projections (bf16 WMMA, async double-buffered)
  dim3 gqkv(D_ / 128, NT_ / 128);
  k_gemm<<<gqkv, 256, 0, stream>>>(xn, wqt, qf, NT_, D_, D_);
  k_gemm<<<gqkv, 256, 0, stream>>>(xn, wkt, kf_, NT_, D_, D_);
  k_gemm<<<gqkv, 256, 0, stream>>>(xn, wvt, vf, NT_, D_, D_);

  // 5) RoPE -> bf16; V -> transposed bf16 (B,H,HD,S)
  dim3 grot(NT_, H_);
  k_rotary<<<grot, 64, 0, stream>>>(qf, fr, qb);
  k_rotary<<<grot, 64, 0, stream>>>(kf_, fr, kb);
  k_cvt_vt<<<dim3(HD_ / 32, S_ / 32, B_ * H_), 256, 0, stream>>>(vf, vt);

  // 6) flash attention
  k_attn<<<dim3(S_ / 128, H_, B_), 256, 0, stream>>>(qb, kb, vt, maskp, ob);

  // 7) output projection + residual + ffn RMSNorm
  k_gemm<<<gqkv, 256, 0, stream>>>(ob, wot, aproj, NT_, D_, D_);
  k_resid_norm<<<NT_, 256, 0, stream>>>(maskedp, aproj, fnw, hn);

  // 8) fused SwiGLU gate, then down projection into d_out
  k_ffn_gate<<<dim3(HID_ / 64, NT_ / 128), 256, 0, stream>>>(hn, w1t, w3t, a2);
  k_gemm<<<dim3(D_ / 128, NT_ / 128), 256, 0, stream>>>(a2, w2t, out_p, NT_, D_, HID_);
}
